// ContrastiveLoss_5669356836480
// MI455X (gfx1250) — compile-verified, matrix operand order checked by reading, not run on hardware
//
#include <hip/hip_runtime.h>

typedef __attribute__((ext_vector_type(16))) __bf16 v16bf;
typedef __attribute__((ext_vector_type(8)))  float  v8f;
typedef __attribute__((ext_vector_type(4)))  int    v4i;

#define AS1 __attribute__((address_space(1)))
#define AS3 __attribute__((address_space(3)))

#define MARGIN 0.3f
#define TM     128          // block tile rows
#define TN     256          // block tile cols
#define KCU    16           // uints (bf16 pairs) per K chunk (= 32 bf16 = one WMMA K step)
#define LPAD   20           // LDS row pitch in uints (16 data + 4 pad; 16B aligned, rotates banks)

// ---------- async global -> LDS copy (CDNA5, ASYNCcnt-tracked) --------------
#if __has_builtin(__builtin_amdgcn_global_load_async_to_lds_b128)
#define ASYNC_B128(gsrc, ldst)                                                 \
    __builtin_amdgcn_global_load_async_to_lds_b128(                            \
        (AS1 v4i*)(uintptr_t)(gsrc),                                           \
        (AS3 v4i*)(void*)(ldst), 0, 0)
#else
#define ASYNC_B128(gsrc, ldst)                                                 \
    asm volatile("global_load_async_to_lds_b128 %0, %1, %2"                    \
                 :: "v"((unsigned)(uintptr_t)(ldst)),                          \
                    "v"((unsigned)((uintptr_t)(gsrc) & 0xffffffffu)),          \
                    "s"((const void*)((uintptr_t)(gsrc) & ~0xffffffffull))     \
                 : "memory")
#endif

#if __has_builtin(__builtin_amdgcn_s_wait_asynccnt)
#define WAIT_ASYNC0() __builtin_amdgcn_s_wait_asynccnt(0)
#else
#define WAIT_ASYNC0() asm volatile("s_wait_asynccnt 0x0" ::: "memory")
#endif

// ---- fp32 -> packed bf16 (RNE) ---------------------------------------------
__device__ __forceinline__ unsigned pack_bf16(float lo, float hi) {
    unsigned a = __float_as_uint(lo);
    unsigned b = __float_as_uint(hi);
    a += 0x7fffu + ((a >> 16) & 1u);
    b += 0x7fffu + ((b >> 16) & 1u);
    return (a >> 16) | (b & 0xffff0000u);
}

// Convert X (fp32, row-major n x d) into packed-bf16 workspace; also zero d_out.
__global__ void cvt_bf16_kernel(const float* __restrict__ x,
                                unsigned* __restrict__ ws,
                                float* __restrict__ out) {
    if (blockIdx.x == 0 && threadIdx.x == 0) out[0] = 0.0f;
    size_t i = (size_t)blockIdx.x * blockDim.x + threadIdx.x;   // 8 floats per thread
    const float4* x4 = (const float4*)x;
    float4 a = x4[2 * i + 0];
    float4 b = x4[2 * i + 1];
    uint4 o;
    o.x = pack_bf16(a.x, a.y);
    o.y = pack_bf16(a.z, a.w);
    o.z = pack_bf16(b.x, b.y);
    o.w = pack_bf16(b.z, b.w);
    ((uint4*)ws)[i] = o;
}

// ---- main tiled symmetric GEMM + hinge loss --------------------------------
// Block tile: 128 (i) x 256 (j). 8 wave32 in a 2x4 grid, each wave owns 64x64.
// Coverage: keep blocks with bj2 >= bi/2; per-element weight 2 (j>i), 1 (j==i), 0 (j<i).
__global__ __launch_bounds__(256) void contrastive_wmma_kernel(
        const unsigned* __restrict__ xb,   // packed bf16, row pitch dU uints
        const int*      __restrict__ tgt,
        float*          __restrict__ out,
        int dU, int T2, float inv_n) {

    __shared__ unsigned lA[2][TM][LPAD];   // 2 x 10 KB
    __shared__ unsigned lB[2][TN][LPAD];   // 2 x 20 KB
    __shared__ int   lti[TM];
    __shared__ int   ltj[TN];
    __shared__ float red[8];

    const int t    = threadIdx.x;
    const int lane = t & 31;
    const int wid  = t >> 5;           // 0..7 (wave32)
    const int wm   = wid >> 2;         // 0..1 : 64-row band
    const int wn   = wid & 3;          // 0..3 : 64-col band
    const int half = lane >> 4;        // K-half select per WMMA layout
    const int l15  = lane & 15;

    // block decode: keep (bi, bj2) with bj2 >= bi>>1
    int bid = blockIdx.x, bi = 0, rl = T2;
    while (bid >= rl) { bid -= rl; bi++; rl = T2 - (bi >> 1); }
    const int bj2 = (bi >> 1) + bid;
    const int i_base = bi * TM;
    const int j_base = bj2 * TN;

    ltj[t] = tgt[j_base + t];
    if (t < TM) lti[t] = tgt[i_base + t];

    v8f acc[4][4];
#pragma unroll
    for (int ms = 0; ms < 4; ++ms)
#pragma unroll
        for (int ns = 0; ns < 4; ++ns) acc[ms][ns] = (v8f)0.0f;

    const int chunks = dU / KCU;                 // K chunks of 32 bf16
    const unsigned* Ag = xb + (size_t)i_base * dU;
    const unsigned* Bg = xb + (size_t)j_base * dU;

    // staging slots: A = 512 uint4 (2/thread), B = 1024 uint4 (4/thread)
    const int rA0 = t >> 2,          cS = (t & 3) * 4;
    const int rA1 = (256 + t) >> 2;

    auto issue_chunk = [&](int nb, int kc) {
        const int ko = kc * KCU;
        ASYNC_B128(Ag + (size_t)rA0 * dU + ko + cS, &lA[nb][rA0][cS]);
        ASYNC_B128(Ag + (size_t)rA1 * dU + ko + cS, &lA[nb][rA1][cS]);
#pragma unroll
        for (int k = 0; k < 4; ++k) {
            int r = (k * 256 + t) >> 2;
            ASYNC_B128(Bg + (size_t)r * dU + ko + cS, &lB[nb][r][cS]);
        }
    };

    union FragU { uint4 q[2]; v16bf v; };

    // prologue: fill buffer 0
    issue_chunk(0, 0);
    WAIT_ASYNC0();
    __syncthreads();

    const int clo = half * 4;                // uint col of low K-run
    const int chi = clo + 8;                 // uint col of high K-run

    for (int kc = 0; kc < chunks; ++kc) {
        const int cur = kc & 1;
        if (kc + 1 < chunks) issue_chunk(cur ^ 1, kc + 1);

        FragU aF[4];
#pragma unroll
        for (int ms = 0; ms < 4; ++ms) {
            int r = wm * 64 + ms * 16 + l15;
            aF[ms].q[0] = *(const uint4*)&lA[cur][r][clo];
            aF[ms].q[1] = *(const uint4*)&lA[cur][r][chi];
        }
#pragma unroll
        for (int ns = 0; ns < 4; ++ns) {
            FragU bF;
            int r = wn * 64 + ns * 16 + l15;
            bF.q[0] = *(const uint4*)&lB[cur][r][clo];
            bF.q[1] = *(const uint4*)&lB[cur][r][chi];
#pragma unroll
            for (int ms = 0; ms < 4; ++ms)
                acc[ms][ns] = __builtin_amdgcn_wmma_f32_16x16x32_bf16(
                    false, aF[ms].v, false, bF.v,
                    (short)0, acc[ms][ns], false, false);
        }

        WAIT_ASYNC0();            // next buffer filled (this wave's copies)
        __syncthreads();          // all waves done reading cur + filling nxt
    }

    // epilogue: hinge terms with per-element symmetry weight
    float lsum = 0.0f;
#pragma unroll
    for (int ms = 0; ms < 4; ++ms) {
#pragma unroll
        for (int ns = 0; ns < 4; ++ns) {
            const int jloc = wn * 64 + ns * 16 + l15;
            const int j    = j_base + jloc;
            const int tj   = ltj[jloc];
#pragma unroll
            for (int v = 0; v < 8; ++v) {
                const int iloc = wm * 64 + ms * 16 + v + (half << 3);
                const int i    = i_base + iloc;
                const int ti   = lti[iloc];
                float s = acc[ms][ns][v];
                float c = (ti == tj) ? ((s < 1.0f)   ? (1.0f - s) : 0.0f)
                                     : ((s > MARGIN) ? s          : 0.0f);
                float we = (j > i) ? 2.0f : ((j == i) ? 1.0f : 0.0f);
                lsum += we * c;
            }
        }
    }

    // wave32 reduction + cross-wave reduction + one atomic per block
#pragma unroll
    for (int off = 16; off > 0; off >>= 1) lsum += __shfl_down(lsum, off, 32);
    if (lane == 0) red[wid] = lsum;
    __syncthreads();
    if (t == 0) {
        float tot = 0.0f;
#pragma unroll
        for (int i = 0; i < 8; ++i) tot += red[i];
        unsafeAtomicAdd(out, tot * inv_n);
    }
}

extern "C" void kernel_launch(void* const* d_in, const int* in_sizes, int n_in,
                              void* d_out, int out_size, void* d_ws, size_t ws_size,
                              hipStream_t stream) {
    const float* x   = (const float*)d_in[0];
    const int*   tgt = (const int*)d_in[1];
    float*       out = (float*)d_out;

    const int n  = in_sizes[1];          // 8192
    const int d  = in_sizes[0] / n;      // 512
    const int dU = d / 2;                // row pitch in packed-bf16 uints
    unsigned* ws = (unsigned*)d_ws;      // needs n*d*2 bytes (8 MB)

    // 1) fp32 -> packed bf16 (+ zero the output accumulator)
    const size_t total  = (size_t)n * d;
    const int cvtBlocks = (int)(total / (256u * 8u));   // 8 floats per thread
    hipLaunchKernelGGL(cvt_bf16_kernel, dim3(cvtBlocks), dim3(256), 0, stream,
                       x, ws, out);

    // 2) diagonal-and-above 128x256 tiles of sim = X X^T, fused hinge loss
    const int T2 = n / TN;               // 32
    int nb = 0;
    for (int bi = 0; bi < n / TM; ++bi) nb += T2 - (bi >> 1);   // 1056 blocks
    hipLaunchKernelGGL(contrastive_wmma_kernel, dim3(nb), dim3(256), 0, stream,
                       ws, tgt, out, dU, T2, 1.0f / (float)n);
}
